// FP8ScaleLinear_19851338842421
// MI455X (gfx1250) — 8-bit matrix operand order repaired and checked by reading, compiled, not run on hardware
//
#include <hip/hip_runtime.h>
#include <hip/hip_bf16.h>

typedef __attribute__((ext_vector_type(16))) int   v16i;
typedef __attribute__((ext_vector_type(8)))  float v8f;
typedef __attribute__((ext_vector_type(4)))  int   i4vec;

#define FP8_MAX 448.0f

#if __has_builtin(__builtin_amdgcn_global_load_async_to_lds_b128) && \
    __has_builtin(__builtin_amdgcn_s_wait_asynccnt)
#define USE_ASYNC_LDS 1
typedef __attribute__((address_space(1))) i4vec GSrcT;
typedef __attribute__((address_space(3))) i4vec LDstT;
#else
#define USE_ASYNC_LDS 0
#endif

__device__ __forceinline__ float bf16_round(float v) {
    unsigned u = __float_as_uint(v);
    unsigned r = u + 0x7FFFu + ((u >> 16) & 1u);
    return __uint_as_float(r & 0xFFFF0000u);
}

__device__ __forceinline__ unsigned char cvt_f32_e4m3_sw(float x) {
    unsigned bits = __float_as_uint(x);
    unsigned s = (bits >> 24) & 0x80u;
    int exp = (int)((bits >> 23) & 0xFF) - 127;
    unsigned man = bits & 0x7FFFFFu;
    if (exp < -9) return (unsigned char)s;
    if (exp < -6) {
        int drop = 20 + (-6 - exp);
        unsigned full = man | 0x800000u;
        unsigned mant = full >> drop;
        unsigned rem  = full & ((1u << drop) - 1u);
        unsigned half = 1u << (drop - 1);
        if (rem > half || (rem == half && (mant & 1u))) mant++;
        if (mant >= 8u) return (unsigned char)(s | 0x08u);
        return (unsigned char)(s | mant);
    }
    unsigned mant = man >> 20;
    unsigned rem  = man & 0xFFFFFu;
    if (rem > 0x80000u || (rem == 0x80000u && (mant & 1u))) {
        mant++;
        if (mant == 8u) { mant = 0u; exp++; }
    }
    if (exp > 8) { exp = 8; mant = 7u; }
    return (unsigned char)(s | (unsigned)((exp + 7) << 3) | mant);
}

__device__ __forceinline__ unsigned pack4_fp8(float a, float b, float c, float d) {
#if __has_builtin(__builtin_amdgcn_cvt_pk_fp8_f32)
    int p = __builtin_amdgcn_cvt_pk_fp8_f32(a, b, 0, false);
    p     = __builtin_amdgcn_cvt_pk_fp8_f32(c, d, p, true);
    return (unsigned)p;
#else
    return (unsigned)cvt_f32_e4m3_sw(a) |
           ((unsigned)cvt_f32_e4m3_sw(b) << 8) |
           ((unsigned)cvt_f32_e4m3_sw(c) << 16) |
           ((unsigned)cvt_f32_e4m3_sw(d) << 24);
#endif
}

__global__ void init_amax_kernel(unsigned* amax) {
    amax[0] = 0u;
    amax[1] = 0u;
}

__global__ __launch_bounds__(256) void amax_kernel(const float4* __restrict__ x,
                                                   int n4, unsigned* __restrict__ out) {
    float m = 0.0f;
    for (int i = blockIdx.x * blockDim.x + threadIdx.x; i < n4;
         i += gridDim.x * blockDim.x) {
        float4 v = x[i];
        m = fmaxf(m, fmaxf(fmaxf(fabsf(v.x), fabsf(v.y)),
                           fmaxf(fabsf(v.z), fabsf(v.w))));
    }
    __shared__ float red[256];
    red[threadIdx.x] = m;
    __syncthreads();
    for (int s = 128; s > 0; s >>= 1) {
        if ((int)threadIdx.x < s)
            red[threadIdx.x] = fmaxf(red[threadIdx.x], red[threadIdx.x + s]);
        __syncthreads();
    }
    if (threadIdx.x == 0)
        atomicMax(out, __float_as_uint(red[0]));
}

__global__ __launch_bounds__(256) void quant_kernel(const float4* __restrict__ x,
                                                    unsigned* __restrict__ q, int n4,
                                                    const unsigned* __restrict__ amax_bits) {
    float amax  = fmaxf(__uint_as_float(*amax_bits), 1e-12f);
    float scale = fminf(FP8_MAX / amax, FP8_MAX);
    for (int i = blockIdx.x * blockDim.x + threadIdx.x; i < n4;
         i += gridDim.x * blockDim.x) {
        float4 v = x[i];
        float a = fminf(fmaxf(v.x * scale, -FP8_MAX), FP8_MAX);
        float b = fminf(fmaxf(v.y * scale, -FP8_MAX), FP8_MAX);
        float c = fminf(fmaxf(v.z * scale, -FP8_MAX), FP8_MAX);
        float d = fminf(fmaxf(v.w * scale, -FP8_MAX), FP8_MAX);
        q[i] = pack4_fp8(a, b, c, d);
    }
}

#define LDS_STRIDE 144

union FragA { v16i v; unsigned long long q[8]; };
union FragB { v16i v; uint4 d4[4]; };

__global__ __launch_bounds__(256, 1) void gemm_fp8_kernel(
        const unsigned char* __restrict__ xq,
        const unsigned char* __restrict__ wq,
        const float* __restrict__ bias,
        float* __restrict__ out,
        const unsigned* __restrict__ amaxX,
        const unsigned* __restrict__ amaxW,
        int K, int N) {

    __shared__ __align__(16) unsigned char sA[2][128 * LDS_STRIDE];
    __shared__ __align__(16) unsigned char sB[2][128 * LDS_STRIDE];

    const int t      = threadIdx.x;
    const int lane   = t & 31;
    const int waveId = t >> 5;
    const int waveM  = waveId & 1;
    const int waveN  = waveId >> 1;
    const int r16    = lane & 15;
    const int half   = lane >> 4;

    const int mBase = blockIdx.x * 128;
    const int nBase = blockIdx.y * 128;

    int rowC[4], offC[4];
#pragma unroll
    for (int i = 0; i < 4; ++i) {
        int chunk = t + i * 256;
        rowC[i] = chunk >> 3;
        offC[i] = (chunk & 7) * 16;
    }

    float ax = fmaxf(__uint_as_float(*amaxX), 1e-12f);
    float aw = fmaxf(__uint_as_float(*amaxW), 1e-12f);
    float dscale = (1.0f / fminf(FP8_MAX / ax, FP8_MAX)) *
                   (1.0f / fminf(FP8_MAX / aw, FP8_MAX));

    float biasv[2];
#pragma unroll
    for (int tn = 0; tn < 2; ++tn)
        biasv[tn] = bias[nBase + waveN * 32 + tn * 16 + r16];

    auto load_tile = [&](int buf, int k0) {
#if USE_ASYNC_LDS
#pragma unroll
        for (int i = 0; i < 4; ++i) {
            __builtin_amdgcn_global_load_async_to_lds_b128(
                (GSrcT*)(void*)(xq + (size_t)(mBase + rowC[i]) * K + k0 + offC[i]),
                (LDstT*)(&sA[buf][rowC[i] * LDS_STRIDE + offC[i]]), 0, 0);
            __builtin_amdgcn_global_load_async_to_lds_b128(
                (GSrcT*)(void*)(wq + (size_t)(nBase + rowC[i]) * K + k0 + offC[i]),
                (LDstT*)(&sB[buf][rowC[i] * LDS_STRIDE + offC[i]]), 0, 0);
        }
#else
        uint4 aReg[4], bReg[4];
#pragma unroll
        for (int i = 0; i < 4; ++i) {
            aReg[i] = *(const uint4*)(xq + (size_t)(mBase + rowC[i]) * K + k0 + offC[i]);
            bReg[i] = *(const uint4*)(wq + (size_t)(nBase + rowC[i]) * K + k0 + offC[i]);
        }
#pragma unroll
        for (int i = 0; i < 4; ++i) {
            *(uint4*)(&sA[buf][rowC[i] * LDS_STRIDE + offC[i]]) = aReg[i];
            *(uint4*)(&sB[buf][rowC[i] * LDS_STRIDE + offC[i]]) = bReg[i];
        }
#endif
    };
    auto wait_async = [&]() {
#if USE_ASYNC_LDS
        __builtin_amdgcn_s_wait_asynccnt(0);
#endif
    };

    v8f acc[4][2] = {};

    load_tile(0, 0);
    wait_async();
    __syncthreads();

    const int KT = K >> 7;
    for (int it = 0; it < KT; ++it) {
        const int cur = it & 1;
        if (it + 1 < KT)
            load_tile((it + 1) & 1, (it + 1) << 7);

        FragA a[4];
#pragma unroll
        for (int tm = 0; tm < 4; ++tm) {
            const unsigned char* base =
                &sA[cur][(waveM * 64 + tm * 16 + r16) * LDS_STRIDE] + half * 16;
#pragma unroll
            for (int blk = 0; blk < 2; ++blk)
#pragma unroll
                for (int c = 0; c < 4; ++c)
                    a[tm].q[blk * 4 + c] =
                        *(const unsigned long long*)(base + blk * 64 + (c >> 1) * 32 + (c & 1) * 8);
        }

        FragB b[2];
#pragma unroll
        for (int tn = 0; tn < 2; ++tn) {
            const unsigned char* base =
                &sB[cur][(waveN * 32 + tn * 16 + r16) * LDS_STRIDE] + half * 16;
#pragma unroll
            for (int c = 0; c < 4; ++c)
                b[tn].d4[c] = *(const uint4*)(base + c * 32);
        }

#pragma unroll
        for (int tm = 0; tm < 4; ++tm)
#pragma unroll
            for (int tn = 0; tn < 2; ++tn)
                acc[tm][tn] = __builtin_amdgcn_wmma_f32_16x16x128_fp8_fp8(
                    a[tm].v, b[tn].v, (short)0, acc[tm][tn], false, false);

        wait_async();
        __syncthreads();
    }

#pragma unroll
    for (int tm = 0; tm < 4; ++tm) {
        int mg = mBase + waveM * 64 + tm * 16 + half * 8;
#pragma unroll
        for (int tn = 0; tn < 2; ++tn) {
            int ng = nBase + waveN * 32 + tn * 16 + r16;
            float bv = biasv[tn];
#pragma unroll
            for (int j = 0; j < 8; ++j) {
                float v = acc[tm][tn][j] * dscale + bv;
                out[(size_t)(mg + j) * N + ng] = bf16_round(v);
            }
        }
    }
}

extern "C" void kernel_launch(void* const* d_in, const int* in_sizes, int n_in,
                              void* d_out, int out_size, void* d_ws, size_t ws_size,
                              hipStream_t stream) {
    const float* x    = (const float*)d_in[0];
    const float* w    = (const float*)d_in[1];
    const float* bias = (const float*)d_in[2];
    float*       out  = (float*)d_out;

    const int IN  = 1024;
    const int OUT = in_sizes[2];
    const long T  = (long)in_sizes[0] / IN;

    unsigned char* ws    = (unsigned char*)d_ws;
    unsigned*      amaxX = (unsigned*)ws;
    unsigned*      amaxW = amaxX + 1;
    unsigned char* xq    = ws + 256;
    unsigned char* wq    = xq + (size_t)T * IN;

    const int n4x = (int)((T * IN) / 4);
    const int n4w = (OUT * IN) / 4;

    init_amax_kernel<<<1, 1, 0, stream>>>(amaxX);
    amax_kernel<<<4096, 256, 0, stream>>>((const float4*)x, n4x, amaxX);
    amax_kernel<<<512, 256, 0, stream>>>((const float4*)w, n4w, amaxW);
    quant_kernel<<<8192, 256, 0, stream>>>((const float4*)x, (unsigned*)xq, n4x, amaxX);
    quant_kernel<<<1024, 256, 0, stream>>>((const float4*)w, (unsigned*)wq, n4w, amaxW);

    dim3 grid((unsigned)(T / 128), (unsigned)(OUT / 128));
    gemm_fp8_kernel<<<grid, 256, 0, stream>>>(xq, wq, bias, out, amaxX, amaxW, IN, OUT);
}
